// LocalCausalAttentionBlock_1906965479938
// MI455X (gfx1250) — matrix operand order repaired and checked by reading, not compiled, not run on hardware
//
#include <hip/hip_runtime.h>
#include <math.h>

typedef __attribute__((ext_vector_type(16))) __bf16 v16bf;
typedef __attribute__((ext_vector_type(8)))  float  v8f;
typedef __attribute__((ext_vector_type(4)))  int    v4i;
typedef __attribute__((address_space(1))) v4i v4i_as1;
typedef __attribute__((address_space(3))) v4i v4i_as3;

#define DIM_   1024
#define S_     2048
#define B_     2
#define M_     (B_*S_)      /* 4096 rows */
#define HEADS_ 16
#define HD_    64
#define WIN_   5

// ---- gfx1250 async global->LDS path (guarded; falls back to sync copy) ----
#if defined(__has_builtin)
#if __has_builtin(__builtin_amdgcn_global_load_async_to_lds_b128)
#define HAVE_ASYNC_LDS 1
#endif
#endif
#ifndef HAVE_ASYNC_LDS
#define HAVE_ASYNC_LDS 0
#endif

__device__ __forceinline__ void cp16_async(const unsigned short* g, unsigned short* l) {
#if HAVE_ASYNC_LDS
  __builtin_amdgcn_global_load_async_to_lds_b128(
      (v4i_as1*)(g), (v4i_as3*)(l), 0, 0);
#else
  *(uint4*)l = *(const uint4*)g;
#endif
}

__device__ __forceinline__ void wait_async_le3() {
#if HAVE_ASYNC_LDS
#if __has_builtin(__builtin_amdgcn_s_wait_asynccnt)
  __builtin_amdgcn_s_wait_asynccnt(3);
#else
  asm volatile("s_wait_asynccnt 3" ::: "memory");
#endif
#endif
}
__device__ __forceinline__ void wait_async_le0() {
#if HAVE_ASYNC_LDS
#if __has_builtin(__builtin_amdgcn_s_wait_asynccnt)
  __builtin_amdgcn_s_wait_asynccnt(0);
#else
  asm volatile("s_wait_asynccnt 0" ::: "memory");
#endif
#endif
}

__device__ __forceinline__ unsigned short f2bf(float f) {
  unsigned int u = __float_as_uint(f);
  unsigned int r = (u + 0x7FFFu + ((u >> 16) & 1u)) >> 16;   // RNE
  return (unsigned short)r;
}
__device__ __forceinline__ float bf2f(unsigned short h) {
  return __uint_as_float(((unsigned int)h) << 16);
}

// ---------------------------------------------------------------- fp32 -> bf16
__global__ __launch_bounds__(256) void cvt_kernel(const float* __restrict__ src,
                                                  unsigned short* __restrict__ dst,
                                                  int n) {
  for (int i = blockIdx.x * 256 + threadIdx.x; i < n; i += gridDim.x * 256)
    dst[i] = f2bf(src[i]);
}

// ---------------------------------------------------------------- bf16 WMMA GEMM
// C[M,N] = A[M,K] @ W[N,K]^T + bias ; optional exact-GELU ; fp32 and/or bf16 out.
// Block = 256 threads = 8 waves; block tile 64(M) x 128(N); K-step 32.
// Double-buffered LDS fed by async global->LDS copies (ASYNCcnt), so the
// next K-slab's HBM traffic overlaps the current slab's WMMAs.
union FragU { uint4 u[2]; v16bf v; };

__global__ __launch_bounds__(256) void gemm_bf16_kernel(
    const unsigned short* __restrict__ A,   // [M,K] bf16 row-major
    const unsigned short* __restrict__ W,   // [N,K] bf16 row-major
    const float* __restrict__ bias,         // [N]
    float* __restrict__ outF,               // [M,N] fp32 (optional)
    unsigned short* __restrict__ outB,      // [M,N] bf16 (optional)
    int M, int N, int K, int act)
{
  __shared__ __align__(16) unsigned short lA[2][64 * 32];
  __shared__ __align__(16) unsigned short lB[2][128 * 32];

  const int tid  = threadIdx.x;
  const int lane = tid & 31;          // wave32
  const int wid  = tid >> 5;          // 0..7
  const int rowB = blockIdx.y * 64;
  const int colB = blockIdx.x * 128;
  const int mW = (wid & 3) * 16;      // wave's M sub-tile
  const int nW = (wid >> 2) * 64;     // wave's N strip (4 x 16)
  const int lm = lane & 15;
  const int kg = lane >> 4;           // lane K-group (ISA 16-bit A/B layout)
  const int ar = tid >> 2, ac = tid & 3;

  // per-thread source pointers (k0 added per step)
  const unsigned short* gA  = A + (size_t)(rowB + ar) * K + ac * 8;
  const unsigned short* gB0 = W + (size_t)(colB + ar) * K + ac * 8;
  const unsigned short* gB1 = W + (size_t)(colB + 64 + ar) * K + ac * 8;
  const int dOff = ar * 32 + ac * 8;

  v8f acc[4] = {};

  // prologue: stage slab 0 into buffer 0
  cp16_async(gA,  &lA[0][dOff]);
  cp16_async(gB0, &lB[0][dOff]);
  cp16_async(gB1, &lB[0][(64 + ar) * 32 + ac * 8]);

  int cur = 0;
  for (int k0 = 0; k0 < K; k0 += 32) {
    if (k0 + 32 < K) {
      // stage next slab into the other buffer while we compute this one
      cp16_async(gA  + k0 + 32, &lA[cur ^ 1][dOff]);
      cp16_async(gB0 + k0 + 32, &lB[cur ^ 1][dOff]);
      cp16_async(gB1 + k0 + 32, &lB[cur ^ 1][(64 + ar) * 32 + ac * 8]);
      wait_async_le3();               // current slab landed; next still in flight
    } else {
      wait_async_le0();               // drain for the final slab
    }
    __syncthreads();

    // A fragment 16x32: lane m = lm, K = kg*8 + e (e<8), 16 + kg*8 + (e-8)
    FragU fa;
    const unsigned short* arow = &lA[cur][(mW + lm) * 32];
    fa.u[0] = *(const uint4*)(arow + kg * 8);
    fa.u[1] = *(const uint4*)(arow + 16 + kg * 8);

#pragma unroll
    for (int t = 0; t < 4; ++t) {
      // B fragment 32x16: lane N = lm, K = kg*8 + e (e<8), 16 + kg*8 + (e-8)
      FragU fb;
      const unsigned short* brow = &lB[cur][(nW + t * 16 + lm) * 32];
      fb.u[0] = *(const uint4*)(brow + kg * 8);
      fb.u[1] = *(const uint4*)(brow + 16 + kg * 8);
      acc[t] = __builtin_amdgcn_wmma_f32_16x16x32_bf16(
          false, fa.v, false, fb.v, (short)0, acc[t], false, false);
    }
    __syncthreads();                  // close reads before buffer reuse
    cur ^= 1;
  }

  // epilogue: C layout — lane 0-15: M=r, lane 16-31: M=8+r; N = lane&15
  const int lmrow = (lane >> 4) * 8;
#pragma unroll
  for (int t = 0; t < 4; ++t) {
    const int col = colB + nW + t * 16 + lm;
    const float bv = bias[col];
#pragma unroll
    for (int r = 0; r < 8; ++r) {
      const int row = rowB + mW + lmrow + r;
      float v = acc[t][r] + bv;
      if (act) v = 0.5f * v * (1.0f + erff(v * 0.70710678118654752f));
      const size_t off = (size_t)row * N + col;
      if (outF) outF[off] = v;
      if (outB) outB[off] = f2bf(v);
    }
  }
}

// ---------------------------------------------------------------- window-5 attention
// one wave per (b, s, h); hd=64 -> 2 elems/lane; qkv bf16 [M, 3*DIM]
__global__ __launch_bounds__(256) void attn_kernel(
    const unsigned short* __restrict__ qkv,
    unsigned short* __restrict__ outb)
{
  const int wid  = blockIdx.x * 8 + (threadIdx.x >> 5);
  const int lane = threadIdx.x & 31;
  const int h = wid & (HEADS_ - 1);
  const int m = wid >> 4;            // b*S + s
  const int b = m >> 11;
  const int s = m & (S_ - 1);

  const size_t qoff = (size_t)m * (3 * DIM_) + h * HD_ + 2 * lane;
  const float q0 = bf2f(qkv[qoff]);
  const float q1 = bf2f(qkv[qoff + 1]);

  const int j0  = (s >= WIN_ - 1) ? s - (WIN_ - 1) : 0;
  const int cnt = s - j0 + 1;

  float sc[WIN_];
#pragma unroll
  for (int w = 0; w < WIN_; ++w) {
    float p = 0.0f;
    if (w < cnt) {
      const size_t kr = (size_t)(b * S_ + j0 + w) * (3 * DIM_) + DIM_ + h * HD_ + 2 * lane;
      p = q0 * bf2f(qkv[kr]) + q1 * bf2f(qkv[kr + 1]);
    }
#pragma unroll
    for (int o = 16; o > 0; o >>= 1) p += __shfl_xor(p, o, 32);
    sc[w] = p * 0.125f;              // 1/sqrt(64)
  }

  float mx = -3.4e38f;
  for (int w = 0; w < cnt; ++w) mx = fmaxf(mx, sc[w]);
  float pr[WIN_], den = 0.0f;
  for (int w = 0; w < cnt; ++w) { pr[w] = __expf(sc[w] - mx); den += pr[w]; }
  const float inv = 1.0f / den;

  float o0 = 0.0f, o1 = 0.0f;
  for (int w = 0; w < cnt; ++w) {
    const float pw = pr[w] * inv;
    const size_t vr = (size_t)(b * S_ + j0 + w) * (3 * DIM_) + 2 * DIM_ + h * HD_ + 2 * lane;
    o0 += pw * bf2f(qkv[vr]);
    o1 += pw * bf2f(qkv[vr + 1]);
  }
  const size_t oo = (size_t)m * DIM_ + h * HD_ + 2 * lane;
  outb[oo]     = f2bf(o0);
  outb[oo + 1] = f2bf(o1);
}

// ---------------------------------------------------------------- fused residual + LayerNorm
__global__ __launch_bounds__(256) void ln_kernel(
    const float* __restrict__ r1, const float* __restrict__ r2,
    const float* __restrict__ g,  const float* __restrict__ be,
    float* __restrict__ outF, unsigned short* __restrict__ outB)
{
  const int row  = blockIdx.x;
  const int tid  = threadIdx.x;
  const int lane = tid & 31;
  const int wid  = tid >> 5;
  __shared__ float red[16];

  float v[4], s = 0.0f, s2 = 0.0f;
#pragma unroll
  for (int i = 0; i < 4; ++i) {
    const int c = tid + i * 256;
    const float x = r1[(size_t)row * DIM_ + c] + r2[(size_t)row * DIM_ + c];
    v[i] = x; s += x; s2 += x * x;
  }
#pragma unroll
  for (int o = 16; o > 0; o >>= 1) { s += __shfl_xor(s, o, 32); s2 += __shfl_xor(s2, o, 32); }
  if (lane == 0) { red[wid] = s; red[8 + wid] = s2; }
  __syncthreads();
  s = 0.0f; s2 = 0.0f;
#pragma unroll
  for (int k = 0; k < 8; ++k) { s += red[k]; s2 += red[8 + k]; }

  const float mu  = s  * (1.0f / DIM_);
  const float var = s2 * (1.0f / DIM_) - mu * mu;
  const float rs  = rsqrtf(var + 1e-5f);
#pragma unroll
  for (int i = 0; i < 4; ++i) {
    const int c = tid + i * 256;
    const float y = (v[i] - mu) * rs * g[c] + be[c];
    outF[(size_t)row * DIM_ + c] = y;
    if (outB) outB[(size_t)row * DIM_ + c] = f2bf(y);
  }
}

// ---------------------------------------------------------------- launch
extern "C" void kernel_launch(void* const* d_in, const int* in_sizes, int n_in,
                              void* d_out, int out_size, void* d_ws, size_t ws_size,
                              hipStream_t stream) {
  (void)in_sizes; (void)n_in; (void)out_size; (void)ws_size;
  const float* x     = (const float*)d_in[0];
  const float* w_in  = (const float*)d_in[1];
  const float* b_in  = (const float*)d_in[2];
  const float* w_out = (const float*)d_in[3];
  const float* b_out = (const float*)d_in[4];
  const float* ln1g  = (const float*)d_in[5];
  const float* ln1b  = (const float*)d_in[6];
  const float* ln2g  = (const float*)d_in[7];
  const float* ln2b  = (const float*)d_in[8];
  const float* w1    = (const float*)d_in[9];
  const float* b1    = (const float*)d_in[10];
  const float* w2    = (const float*)d_in[11];
  const float* b2    = (const float*)d_in[12];
  float* out = (float*)d_out;

  // workspace layout (bytes, all 16B aligned)
  char* ws = (char*)d_ws;
  constexpr size_t O_XB   = 0;                                   // x bf16      8 MB
  constexpr size_t O_WIN  = O_XB   + (size_t)M_*DIM_*2;          // w_in bf16   6 MB
  constexpr size_t O_WOUT = O_WIN  + (size_t)3*DIM_*DIM_*2;      // w_out bf16  2 MB
  constexpr size_t O_W1   = O_WOUT + (size_t)DIM_*DIM_*2;        // w1 bf16     8 MB
  constexpr size_t O_W2   = O_W1   + (size_t)4*DIM_*DIM_*2;      // w2 bf16     8 MB
  constexpr size_t O_QKV  = O_W2   + (size_t)DIM_*4*DIM_*2;      // qkv bf16   24 MB
  constexpr size_t O_ATT  = O_QKV  + (size_t)M_*3*DIM_*2;        // attn bf16   8 MB
  constexpr size_t O_PROJ = O_ATT  + (size_t)M_*DIM_*2;          // proj f32   16 MB
  constexpr size_t O_X1F  = O_PROJ + (size_t)M_*DIM_*4;          // x1 f32     16 MB
  constexpr size_t O_X1B  = O_X1F  + (size_t)M_*DIM_*4;          // x1 bf16     8 MB
  constexpr size_t O_H    = O_X1B  + (size_t)M_*DIM_*2;          // h bf16     32 MB
  constexpr size_t O_MLP  = O_H    + (size_t)M_*4*DIM_*2;        // mlp f32    16 MB

  unsigned short* xb    = (unsigned short*)(ws + O_XB);
  unsigned short* winb  = (unsigned short*)(ws + O_WIN);
  unsigned short* woutb = (unsigned short*)(ws + O_WOUT);
  unsigned short* w1b   = (unsigned short*)(ws + O_W1);
  unsigned short* w2b   = (unsigned short*)(ws + O_W2);
  unsigned short* qkvb  = (unsigned short*)(ws + O_QKV);
  unsigned short* attnb = (unsigned short*)(ws + O_ATT);
  float*          projf = (float*)(ws + O_PROJ);
  float*          x1f   = (float*)(ws + O_X1F);
  unsigned short* x1b   = (unsigned short*)(ws + O_X1B);
  unsigned short* hb    = (unsigned short*)(ws + O_H);
  float*          mlpf  = (float*)(ws + O_MLP);

  // fp32 -> bf16 conversions
  cvt_kernel<<<2048, 256, 0, stream>>>(x,     xb,    M_ * DIM_);
  cvt_kernel<<<2048, 256, 0, stream>>>(w_in,  winb,  3 * DIM_ * DIM_);
  cvt_kernel<<<1024, 256, 0, stream>>>(w_out, woutb, DIM_ * DIM_);
  cvt_kernel<<<2048, 256, 0, stream>>>(w1,    w1b,   4 * DIM_ * DIM_);
  cvt_kernel<<<2048, 256, 0, stream>>>(w2,    w2b,   DIM_ * 4 * DIM_);

  // qkv = x @ w_in^T + b_in           (M=4096, N=3072, K=1024)
  gemm_bf16_kernel<<<dim3(3 * DIM_ / 128, M_ / 64), 256, 0, stream>>>(
      xb, winb, b_in, nullptr, qkvb, M_, 3 * DIM_, DIM_, 0);

  // local causal window-5 attention
  attn_kernel<<<M_ * HEADS_ / 8, 256, 0, stream>>>(qkvb, attnb);

  // proj = attn @ w_out^T + b_out     (N=1024, K=1024)
  gemm_bf16_kernel<<<dim3(DIM_ / 128, M_ / 64), 256, 0, stream>>>(
      attnb, woutb, b_out, projf, nullptr, M_, DIM_, DIM_, 0);

  // x1 = LN(x + proj)
  ln_kernel<<<M_, 256, 0, stream>>>(x, projf, ln1g, ln1b, x1f, x1b);

  // h = gelu(x1 @ w1^T + b1)          (N=4096, K=1024)
  gemm_bf16_kernel<<<dim3(4 * DIM_ / 128, M_ / 64), 256, 0, stream>>>(
      x1b, w1b, b1, nullptr, hb, M_, 4 * DIM_, DIM_, 1);

  // mlp = h @ w2^T + b2               (N=1024, K=4096)
  gemm_bf16_kernel<<<dim3(DIM_ / 128, M_ / 64), 256, 0, stream>>>(
      hb, w2b, b2, mlpf, nullptr, M_, DIM_, 4 * DIM_, 0);

  // out = LN(x1 + mlp)
  ln_kernel<<<M_, 256, 0, stream>>>(x1f, mlpf, ln2g, ln2b, out, nullptr);
}
